// CausalSelfAttention_89524298318587
// MI455X (gfx1250) — matrix operand order repaired and checked by reading, not compiled, not run on hardware
//
#include <hip/hip_runtime.h>
#include <hip/hip_bf16.h>

typedef __attribute__((ext_vector_type(16))) _Float16 v16h;
typedef __attribute__((ext_vector_type(8)))  float    v8f;

#define Bc 8
#define Tc 1024
#define Cc 768
#define Hc 12
#define Dc 64
#define Mc (Bc * Tc)        // 8192
#define N1c (3 * Cc)        // 2304
#define Kc Cc               // 768

// Use gfx1250 async global->LDS copies for attention K/V tiles.
#define USE_ASYNC_LDS 1

union F16x16 {
    v16h v;
    unsigned int u[8];
    _Float16 h[16];
};

// ---- WMMA fragment loaders (CDNA5 16-bit layouts, wave32) ----

// A-matrix 16x32 (MxK): lane l holds row l%16; VGPR v holds K pair:
//   K = (v/4)*16 + (l/16)*8 + (v%4)*2  (pair contiguous in row-major A)
__device__ inline v16h loadA16x32(const _Float16* base, int ld) {
    int lane = threadIdx.x & 31;
    int row = lane & 15, half = lane >> 4;
    F16x16 f;
#pragma unroll
    for (int v = 0; v < 8; ++v) {
        int k = ((v >> 2) << 4) + (half << 3) + ((v & 3) << 1);
        f.u[v] = *(const unsigned int*)(base + row * ld + k);
    }
    return f.v;
}

// B-matrix 32x16 stored transposed: element (k,n) = base[n*ld + k].
// lane l holds col l%16; VGPR v holds K pair K = (v/4)*16 + (l/16)*8 + (v%4)*2.
__device__ inline v16h loadB32x16_T(const _Float16* base, int ld) {
    int lane = threadIdx.x & 31;
    int col = lane & 15, half = lane >> 4;
    F16x16 f;
#pragma unroll
    for (int v = 0; v < 8; ++v) {
        int k = ((v >> 2) << 4) + (half << 3) + ((v & 3) << 1);
        f.u[v] = *(const unsigned int*)(base + col * ld + k);
    }
    return f.v;
}

__device__ inline v8f wmma_f16(v16h a, v16h b, v8f c) {
    return __builtin_amdgcn_wmma_f32_16x16x32_f16(false, a, false, b, (short)0, c,
                                                  false, false);
}

__device__ inline float redmax16(float x) {
    x = fmaxf(x, __shfl_xor(x, 1));
    x = fmaxf(x, __shfl_xor(x, 2));
    x = fmaxf(x, __shfl_xor(x, 4));
    x = fmaxf(x, __shfl_xor(x, 8));
    return x;
}
__device__ inline float redsum16(float x) {
    x += __shfl_xor(x, 1);
    x += __shfl_xor(x, 2);
    x += __shfl_xor(x, 4);
    x += __shfl_xor(x, 8);
    return x;
}

#if USE_ASYNC_LDS
// Per-lane async 16B global -> LDS copy (gfx1250): tracked by ASYNCcnt.
__device__ inline void async_copy_b128(const _Float16* gptr, const _Float16* lptr) {
    unsigned lds_off = (unsigned)(size_t)lptr;
    unsigned long long gaddr = (unsigned long long)(size_t)gptr;
    asm volatile("global_load_async_to_lds_b128 %0, %1, off"
                 :
                 : "v"(lds_off), "v"(gaddr)
                 : "memory");
}
__device__ inline void async_wait0() {
    asm volatile("s_wait_asynccnt 0" ::: "memory");
}
#endif

// ---------------- Kernel 0: f32 -> f16 conversions (+ weight transpose) ----
__global__ __launch_bounds__(256) void cvt_kernel(const float* __restrict__ x,
                                                  const float* __restrict__ wqkv,
                                                  const float* __restrict__ wo,
                                                  _Float16* __restrict__ xh,
                                                  _Float16* __restrict__ wqkvt,
                                                  _Float16* __restrict__ wot) {
    size_t i = (size_t)blockIdx.x * blockDim.x + threadIdx.x;
    const size_t NX = (size_t)Mc * Kc;       // 6291456
    const size_t NW1 = (size_t)Kc * N1c;     // 1769472
    const size_t NW2 = (size_t)Kc * Cc;      // 589824
    if (i < NX) xh[i] = (_Float16)x[i];
    if (i < NW1) {
        int kk = (int)(i / N1c), n = (int)(i % N1c);
        wqkvt[(size_t)n * Kc + kk] = (_Float16)wqkv[i];
    }
    if (i < NW2) {
        int kk = (int)(i / Cc), n = (int)(i % Cc);
        wot[(size_t)n * Kc + kk] = (_Float16)wo[i];
    }
}

// ---------------- Kernel 1: QKV GEMM + scatter ----------------
// C[M, 3C] = Xh[M,768] * WqkvT[2304,768]^T + b ; scatter q,k -> [B,H,T,D],
// v -> [B,H,D,T] (transposed for the attention P@V operand).
__global__ __launch_bounds__(256) void gemm_qkv_kernel(
    const _Float16* __restrict__ A, const _Float16* __restrict__ Bt,
    const float* __restrict__ bias, _Float16* __restrict__ qb,
    _Float16* __restrict__ kb, _Float16* __restrict__ vb) {
    __shared__ __align__(16) _Float16 As[128 * 32];
    __shared__ __align__(16) _Float16 Bs[128 * 32];  // [N][K] transposed tile

    const int m0 = blockIdx.x * 128;
    const int n0 = blockIdx.y * 128;
    const int w = threadIdx.x >> 5;   // 0..7
    const int wm = w >> 2;            // 0..1 -> 64 rows
    const int wn = w & 3;             // 0..3 -> 32 cols
    const int lane = threadIdx.x & 31;
    const int colb = lane & 15, half = lane >> 4;

    v8f acc[4][2];
#pragma unroll
    for (int i = 0; i < 4; ++i)
#pragma unroll
        for (int j = 0; j < 2; ++j) acc[i][j] = (v8f){0, 0, 0, 0, 0, 0, 0, 0};

    for (int k0 = 0; k0 < Kc; k0 += 32) {
#pragma unroll
        for (int c = 0; c < 2; ++c) {
            int chunk = threadIdx.x + c * 256;   // 512 chunks of 8 halves
            int ar = chunk >> 2, aw = chunk & 3; // 128 rows x 4 chunks
            *(uint4*)&As[ar * 32 + aw * 8] =
                *(const uint4*)&A[(size_t)(m0 + ar) * Kc + k0 + aw * 8];
            *(uint4*)&Bs[ar * 32 + aw * 8] =
                *(const uint4*)&Bt[(size_t)(n0 + ar) * Kc + k0 + aw * 8];
        }
        __syncthreads();
        v16h af[4], bf[2];
#pragma unroll
        for (int i = 0; i < 4; ++i) af[i] = loadA16x32(&As[(wm * 64 + i * 16) * 32], 32);
#pragma unroll
        for (int j = 0; j < 2; ++j) bf[j] = loadB32x16_T(&Bs[(wn * 32 + j * 16) * 32], 32);
#pragma unroll
        for (int i = 0; i < 4; ++i)
#pragma unroll
            for (int j = 0; j < 2; ++j) acc[i][j] = wmma_f16(af[i], bf[j], acc[i][j]);
        __syncthreads();
    }

    // Epilogue: bias + scatter. 128-row M block never crosses a batch
    // boundary (1024 % 128 == 0) so b is block-uniform; no div/mod needed.
    const int b = m0 >> 10;                    // batch index (uniform)
    const int tb = (m0 & 1023) + wm * 64;      // token base within batch
#pragma unroll
    for (int i = 0; i < 4; ++i) {
#pragma unroll
        for (int j = 0; j < 2; ++j) {
            int n = n0 + wn * 32 + j * 16 + colb;
            float bv = bias[n];
            int which = (n >= 2 * Cc) ? 2 : ((n >= Cc) ? 1 : 0);
            int c = n - which * Cc;
            int h = c >> 6, d = c & 63;
            size_t bh = (size_t)(b * Hc + h);
            if (which == 2) {
                _Float16* p = vb + (bh * Dc + d) * Tc;  // [B,H,D,T]
#pragma unroll
                for (int r = 0; r < 8; ++r) {
                    int t = tb + i * 16 + r + 8 * half;
                    p[t] = (_Float16)(acc[i][j][r] + bv);
                }
            } else {
                _Float16* p = ((which == 0) ? qb : kb) + bh * Tc * Dc + d;
#pragma unroll
                for (int r = 0; r < 8; ++r) {
                    int t = tb + i * 16 + r + 8 * half;
                    p[(size_t)t * Dc] = (_Float16)(acc[i][j][r] + bv);
                }
            }
        }
    }
}

// ---------------- Kernel 2: flash attention ----------------
// grid: (H, B, T/64), block: 128 threads (4 waves, 16 query rows each)
// q,k: [B,H,T,D] f16 ; v: [B,H,D,T] f16 ; o: [B,T,C] f16
__global__ __launch_bounds__(128) void attn_kernel(const _Float16* __restrict__ q,
                                                   const _Float16* __restrict__ k,
                                                   const _Float16* __restrict__ v,
                                                   _Float16* __restrict__ o) {
    __shared__ __align__(16) _Float16 Ks[64 * 64];  // [key][d]
    __shared__ __align__(16) _Float16 Vs[64 * 64];  // [d][key] (from transposed V)
    __shared__ __align__(16) _Float16 Ps[4][16 * 64];

    const int h = blockIdx.x;        // 0..11
    const int b = blockIdx.y;        // 0..7
    const int qb = blockIdx.z;       // 0..15
    const int bh = b * Hc + h;
    const int w = threadIdx.x >> 5;  // 0..3
    const int lane = threadIdx.x & 31;
    const int colb = lane & 15, half = lane >> 4;
    const size_t head = (size_t)bh * Tc * Dc;
    const int q0 = qb * 64 + w * 16;

    v16h qa0 = loadA16x32(q + head + (size_t)q0 * Dc, Dc);
    v16h qa1 = loadA16x32(q + head + (size_t)q0 * Dc + 32, Dc);

    float mrow[8], lrow[8];
    v8f oacc[4];
#pragma unroll
    for (int r = 0; r < 8; ++r) { mrow[r] = -1e30f; lrow[r] = 0.0f; }
#pragma unroll
    for (int df = 0; df < 4; ++df) oacc[df] = (v8f){0, 0, 0, 0, 0, 0, 0, 0};

    for (int kt = 0; kt <= qb; ++kt) {
        const _Float16* Kg = k + head + (size_t)kt * 64 * Dc;  // [key][d]
        const _Float16* Vg = v + head + (size_t)kt * 64;       // [d][t] tile col
#pragma unroll
        for (int c = 0; c < 4; ++c) {
            int chunk = threadIdx.x + c * 128;  // 512 chunks of 8 halves
            int row = chunk >> 3, wi = chunk & 7;
#if USE_ASYNC_LDS
            async_copy_b128(&Kg[row * Dc + wi * 8], &Ks[row * 64 + wi * 8]);
            async_copy_b128(&Vg[(size_t)row * Tc + wi * 8], &Vs[row * 64 + wi * 8]);
#else
            *(uint4*)&Ks[row * 64 + wi * 8] = *(const uint4*)&Kg[row * Dc + wi * 8];
            *(uint4*)&Vs[row * 64 + wi * 8] =
                *(const uint4*)&Vg[(size_t)row * Tc + wi * 8];
#endif
        }
#if USE_ASYNC_LDS
        async_wait0();
#endif
        __syncthreads();

        // S = Q * K^T / sqrt(D): B element (d, key) = Ks[key*64 + d]
        v8f sf[4];
#pragma unroll
        for (int n = 0; n < 4; ++n) {
            v8f z = (v8f){0, 0, 0, 0, 0, 0, 0, 0};
            z = wmma_f16(qa0, loadB32x16_T(&Ks[(n * 16) * 64 + 0], 64), z);
            z = wmma_f16(qa1, loadB32x16_T(&Ks[(n * 16) * 64 + 32], 64), z);
            sf[n] = z;
        }

        // scale + causal mask, per-row stats
        float pmax[8];
#pragma unroll
        for (int r = 0; r < 8; ++r) pmax[r] = -3.0e38f;
#pragma unroll
        for (int n = 0; n < 4; ++n) {
            int key = kt * 64 + n * 16 + colb;
#pragma unroll
            for (int r = 0; r < 8; ++r) {
                int row = q0 + r + 8 * half;
                float x = sf[n][r] * 0.125f;
                if (key > row) x = -3.0e38f;
                sf[n][r] = x;
                pmax[r] = fmaxf(pmax[r], x);
            }
        }
        float alpha[8], psum[8];
#pragma unroll
        for (int r = 0; r < 8; ++r) {
            float tm = redmax16(pmax[r]);
            float mnew = fmaxf(mrow[r], tm);
            alpha[r] = __expf(mrow[r] - mnew);
            mrow[r] = mnew;
            psum[r] = 0.0f;
        }
#pragma unroll
        for (int n = 0; n < 4; ++n) {
#pragma unroll
            for (int r = 0; r < 8; ++r) {
                float p = __expf(sf[n][r] - mrow[r]);
                Ps[w][(r + 8 * half) * 64 + n * 16 + colb] = (_Float16)p;
                psum[r] += p;
            }
        }
#pragma unroll
        for (int r = 0; r < 8; ++r) {
            float ts = redsum16(psum[r]);
            lrow[r] = lrow[r] * alpha[r] + ts;
        }
#pragma unroll
        for (int df = 0; df < 4; ++df)
#pragma unroll
            for (int r = 0; r < 8; ++r) oacc[df][r] = oacc[df][r] * alpha[r];

        // O += P * V: A = P (16x64) in LDS; B element (key, d) = Vs[d*64 + key]
        v16h pa0 = loadA16x32(&Ps[w][0], 64);
        v16h pa1 = loadA16x32(&Ps[w][32], 64);
#pragma unroll
        for (int df = 0; df < 4; ++df) {
            oacc[df] = wmma_f16(pa0, loadB32x16_T(&Vs[(df * 16) * 64 + 0], 64), oacc[df]);
            oacc[df] = wmma_f16(pa1, loadB32x16_T(&Vs[(df * 16) * 64 + 32], 64), oacc[df]);
        }
        __syncthreads();
    }

    // write normalized output to f16 [B, T, C] with c = h*64 + d
#pragma unroll
    for (int df = 0; df < 4; ++df) {
        int d = df * 16 + colb;
#pragma unroll
        for (int r = 0; r < 8; ++r) {
            int t = q0 + r + 8 * half;
            float val = oacc[df][r] / lrow[r];
            o[((size_t)b * Tc + t) * Cc + h * Dc + d] = (_Float16)val;
        }
    }
}

// ---------------- Kernel 3: output projection ----------------
// out[M, 768] (f32) = attn[M, 768] * WoT[768,768]^T + b_o
__global__ __launch_bounds__(256) void gemm_out_kernel(const _Float16* __restrict__ A,
                                                       const _Float16* __restrict__ Bt,
                                                       const float* __restrict__ bias,
                                                       float* __restrict__ out) {
    __shared__ __align__(16) _Float16 As[128 * 32];
    __shared__ __align__(16) _Float16 Bs[128 * 32];

    const int m0 = blockIdx.x * 128;
    const int n0 = blockIdx.y * 128;
    const int w = threadIdx.x >> 5;
    const int wm = w >> 2;
    const int wn = w & 3;
    const int lane = threadIdx.x & 31;
    const int colb = lane & 15, half = lane >> 4;

    v8f acc[4][2];
#pragma unroll
    for (int i = 0; i < 4; ++i)
#pragma unroll
        for (int j = 0; j < 2; ++j) acc[i][j] = (v8f){0, 0, 0, 0, 0, 0, 0, 0};

    for (int k0 = 0; k0 < Kc; k0 += 32) {
#pragma unroll
        for (int c = 0; c < 2; ++c) {
            int chunk = threadIdx.x + c * 256;
            int ar = chunk >> 2, aw = chunk & 3;
            *(uint4*)&As[ar * 32 + aw * 8] =
                *(const uint4*)&A[(size_t)(m0 + ar) * Kc + k0 + aw * 8];
            *(uint4*)&Bs[ar * 32 + aw * 8] =
                *(const uint4*)&Bt[(size_t)(n0 + ar) * Kc + k0 + aw * 8];
        }
        __syncthreads();
        v16h af[4], bf[2];
#pragma unroll
        for (int i = 0; i < 4; ++i) af[i] = loadA16x32(&As[(wm * 64 + i * 16) * 32], 32);
#pragma unroll
        for (int j = 0; j < 2; ++j) bf[j] = loadB32x16_T(&Bs[(wn * 32 + j * 16) * 32], 32);
#pragma unroll
        for (int i = 0; i < 4; ++i)
#pragma unroll
            for (int j = 0; j < 2; ++j) acc[i][j] = wmma_f16(af[i], bf[j], acc[i][j]);
        __syncthreads();
    }

#pragma unroll
    for (int i = 0; i < 4; ++i) {
#pragma unroll
        for (int j = 0; j < 2; ++j) {
            int n = n0 + wn * 32 + j * 16 + colb;
            float bv = bias[n];
#pragma unroll
            for (int r = 0; r < 8; ++r) {
                int row = m0 + wm * 64 + i * 16 + r + 8 * half;
                out[(size_t)row * Cc + n] = acc[i][j][r] + bv;
            }
        }
    }
}

extern "C" void kernel_launch(void* const* d_in, const int* in_sizes, int n_in,
                              void* d_out, int out_size, void* d_ws, size_t ws_size,
                              hipStream_t stream) {
    const float* x = (const float*)d_in[0];
    const float* W_qkv = (const float*)d_in[1];
    const float* b_qkv = (const float*)d_in[2];
    const float* W_o = (const float*)d_in[3];
    const float* b_o = (const float*)d_in[4];
    float* out = (float*)d_out;

    // workspace layout (f16 buffers), 256B aligned offsets
    char* ws = (char*)d_ws;
    size_t off = 0;
    auto take = [&](size_t bytes) {
        char* p = ws + off;
        off += (bytes + 255) & ~(size_t)255;
        return p;
    };
    _Float16* xh = (_Float16*)take((size_t)Mc * Kc * 2);        // 12.6 MB
    _Float16* wqkvt = (_Float16*)take((size_t)Kc * N1c * 2);    // 3.5 MB (transposed)
    _Float16* wot = (_Float16*)take((size_t)Kc * Cc * 2);       // 1.2 MB (transposed)
    _Float16* qh = (_Float16*)take((size_t)Bc * Hc * Tc * Dc * 2);
    _Float16* kh = (_Float16*)take((size_t)Bc * Hc * Tc * Dc * 2);
    _Float16* vh = (_Float16*)take((size_t)Bc * Hc * Tc * Dc * 2);  // [B,H,D,T]
    _Float16* ah = (_Float16*)take((size_t)Mc * Cc * 2);        // attention out f16

    // 0) convert to f16 (+ transpose weights)
    {
        size_t nmax = (size_t)Mc * Kc;
        int blocks = (int)((nmax + 255) / 256);
        cvt_kernel<<<blocks, 256, 0, stream>>>(x, W_qkv, W_o, xh, wqkvt, wot);
    }
    // 1) QKV projection + scatter
    gemm_qkv_kernel<<<dim3(Mc / 128, N1c / 128), 256, 0, stream>>>(xh, wqkvt, b_qkv,
                                                                   qh, kh, vh);
    // 2) flash attention per (head, batch, query-block)
    attn_kernel<<<dim3(Hc, Bc, Tc / 64), 128, 0, stream>>>(qh, kh, vh, ah);
    // 3) output projection (f32 out)
    gemm_out_kernel<<<dim3(Mc / 128, Cc / 128), 256, 0, stream>>>(ah, wot, b_o, out);
}
